// CausalSelfAttention_2207613190871
// MI455X (gfx1250) — matrix operand order repaired and checked by reading, not compiled, not run on hardware
//
#include <hip/hip_runtime.h>
#include <hip/hip_bf16.h>

typedef _Float16 half8  __attribute__((ext_vector_type(8)));
typedef _Float16 half16 __attribute__((ext_vector_type(16)));
typedef float    float8 __attribute__((ext_vector_type(8)));
typedef int      i32x4  __attribute__((ext_vector_type(4)));

constexpr int BSZ = 4, T = 2048, D = 1024, H = 16;
constexpr int BT = BSZ * T;        // 8192 rows
constexpr int NQKV = 3 * D;        // 3072

// ---------------------------------------------------------------------------
// Async global->LDS copy (CDNA5 GLOBAL_LOAD_ASYNC_TO_LDS_B128, ASYNCcnt),
// with a synchronous fallback if this toolchain doesn't declare the builtins.
#if __has_builtin(__builtin_amdgcn_global_load_async_to_lds_b128) && \
    __has_builtin(__builtin_amdgcn_s_wait_asynccnt)
#define HAS_ASYNC_LDS 1
#else
#define HAS_ASYNC_LDS 0
#endif

typedef __attribute__((address_space(1))) i32x4* gas_i32x4_ptr;
typedef __attribute__((address_space(3))) i32x4* las_i32x4_ptr;

__device__ __forceinline__ void cp_b128_to_lds(_Float16* lds, const _Float16* g) {
#if HAS_ASYNC_LDS
    __builtin_amdgcn_global_load_async_to_lds_b128(
        (gas_i32x4_ptr)(uintptr_t)g,
        (las_i32x4_ptr)(uintptr_t)lds, 0, 0);
#else
    *(half8*)lds = *(const half8*)g;
#endif
}
__device__ __forceinline__ void cp_async_wait() {
#if HAS_ASYNC_LDS
    __builtin_amdgcn_s_wait_asynccnt(0);
#endif
}

// ---------------------------------------------------------------------------
// f32 -> f16 convert
__global__ __launch_bounds__(256) void cvt_f16_kernel(const float* __restrict__ in,
                                                      _Float16* __restrict__ out, int n) {
    int i = blockIdx.x * 256 + threadIdx.x;
    if (i < n) out[i] = (_Float16)in[i];
}

// f32 [K][N] -> f16 [N][K] (transpose + convert)
__global__ __launch_bounds__(256) void cvtT_f16_kernel(const float* __restrict__ in,
                                                       _Float16* __restrict__ out, int K, int N) {
    int i = blockIdx.x * 256 + threadIdx.x;
    if (i < K * N) {
        int k = i / N, n = i - k * N;
        out[(size_t)n * K + k] = (_Float16)in[i];
    }
}

// ---------------------------------------------------------------------------
// GEMM: C[M,N] = A[M,K](f16 row-major) * Bt[N,K](f16, B transposed)
// Block: 256 thr = 8 waves (2 M x 4 N). Wave: 32x32 via 2x2 WMMA 16x16 tiles.
// LDS double-buffered; tiles filled with async global->LDS copies.
template <bool OUT_HALF>
__global__ __launch_bounds__(256) void gemm_wmma_kernel(const _Float16* __restrict__ A,
                                                        const _Float16* __restrict__ Bt,
                                                        void* __restrict__ Out,
                                                        int M, int N, int K) {
    __shared__ __align__(16) _Float16 As[2][64 * 40];    // stride 40 halves (bank pad)
    __shared__ __align__(16) _Float16 Bs[2][128 * 40];

    const int tid  = threadIdx.x;
    const int lane = tid & 31;
    const int warp = tid >> 5;
    const int wm = warp >> 2, wn = warp & 3;
    const int n0 = blockIdx.x * 128;
    const int m0 = blockIdx.y * 64;

    auto stage = [&](int buf, int k0) {
        {   // A tile 64x32: 256 x b128
            int row = tid >> 2, cc = (tid & 3) * 8;
            cp_b128_to_lds(&As[buf][row * 40 + cc],
                           A + (size_t)(m0 + row) * K + k0 + cc);
        }
#pragma unroll
        for (int i = 0; i < 2; i++) {   // B tile 128x32: 512 x b128
            int c = tid + 256 * i;
            int row = c >> 2, cc = (c & 3) * 8;
            cp_b128_to_lds(&Bs[buf][row * 40 + cc],
                           Bt + (size_t)(n0 + row) * K + k0 + cc);
        }
    };

    float8 acc[2][2] = {};

    stage(0, 0);
    cp_async_wait();
    __syncthreads();

    for (int k0 = 0; k0 < K; k0 += 32) {
        const int cur = (k0 >> 5) & 1;
        if (k0 + 32 < K) stage(cur ^ 1, k0 + 32);   // fill next buffer (async)

        half16 af[2], bf[2];
#pragma unroll
        for (int i = 0; i < 2; i++) {   // A frag: rows M=lane&15, K split per half-wave
            int row = wm * 32 + i * 16 + (lane & 15);
            int kb = (lane < 16) ? 0 : 8;
            half8 lo = *(const half8*)(&As[cur][row * 40 + kb]);
            half8 hi = *(const half8*)(&As[cur][row * 40 + kb + 16]);
            af[i] = __builtin_shufflevector(lo, hi, 0,1,2,3,4,5,6,7,8,9,10,11,12,13,14,15);
        }
#pragma unroll
        for (int j = 0; j < 2; j++) {   // B frag: col N=lane&15, K split per half-wave
            int row = wn * 32 + j * 16 + (lane & 15);
            int kb = (lane < 16) ? 0 : 8;
            half8 lo = *(const half8*)(&Bs[cur][row * 40 + kb]);
            half8 hi = *(const half8*)(&Bs[cur][row * 40 + kb + 16]);
            bf[j] = __builtin_shufflevector(lo, hi, 0,1,2,3,4,5,6,7,8,9,10,11,12,13,14,15);
        }
#pragma unroll
        for (int i = 0; i < 2; i++)
#pragma unroll
            for (int j = 0; j < 2; j++)
                acc[i][j] = __builtin_amdgcn_wmma_f32_16x16x32_f16(
                    false, af[i], false, bf[j], (short)0, acc[i][j], false, false);

        cp_async_wait();   // our async fills of the next buffer have landed
        __syncthreads();   // everyone done reading cur + done filling next
    }

#pragma unroll
    for (int i = 0; i < 2; i++)
#pragma unroll
        for (int j = 0; j < 2; j++)
#pragma unroll
            for (int r = 0; r < 8; r++) {
                int row = m0 + wm * 32 + i * 16 + ((lane < 16) ? r : r + 8);
                int col = n0 + wn * 32 + j * 16 + (lane & 15);
                if (OUT_HALF)
                    ((_Float16*)Out)[(size_t)row * N + col] = (_Float16)acc[i][j][r];
                else
                    ((float*)Out)[(size_t)row * N + col] = acc[i][j][r];
            }
}

// ---------------------------------------------------------------------------
// Flash attention. Grid: B*H*(T/64) blocks, 128 threads = 4 waves x 16 q-rows.
// qkv layout: [BT][3072] f16; cols [0,1024)=Q, [1024,2048)=K, [2048,3072)=V,
// head slice h*64 .. h*64+63. Output y: [BT][1024] f16.
__global__ __launch_bounds__(128) void attn_wmma_kernel(const _Float16* __restrict__ qkv,
                                                        _Float16* __restrict__ y) {
    __shared__ __align__(16) _Float16 Ks [64 * 72];       // K block, row-major [k][d]
    __shared__ __align__(16) _Float16 Vts[64 * 72];       // V block transposed [d][k]
    __shared__ __align__(16) _Float16 Ps [4 * 16 * 72];   // per-wave P scratch [16][64]

    const int tid  = threadIdx.x;
    const int lane = tid & 31;
    const int warp = tid >> 5;
    const int blk  = blockIdx.x;
    const int qb = blk & 31;            // T/64 = 32 query blocks
    const int h  = (blk >> 5) & 15;
    const int b  = blk >> 9;
    const size_t rb = (size_t)b * T;

    // Q fragments for this wave's 16 rows (scale 1/sqrt(64) folded in)
    half16 aq[2];
    {
        int qrow = qb * 64 + warp * 16 + (lane & 15);
        const _Float16* qp = qkv + (rb + qrow) * NQKV + h * 64;
#pragma unroll
        for (int f = 0; f < 2; f++) {
            int kb = ((lane < 16) ? 0 : 8) + 32 * f;
            half8 lo = *(const half8*)(qp + kb);
            half8 hi = *(const half8*)(qp + kb + 16);
            half16 a = __builtin_shufflevector(lo, hi, 0,1,2,3,4,5,6,7,8,9,10,11,12,13,14,15);
            aq[f] = a * (_Float16)0.125f;
        }
    }

    float  m[8], lsum[8];
    float8 oacc[4] = {};
#pragma unroll
    for (int r = 0; r < 8; r++) { m[r] = -1e30f; lsum[r] = 0.0f; }

    for (int kbi = 0; kbi <= qb; kbi++) {
        __syncthreads();                 // previous block's LDS reads are done
        // ---- stage K block [64 k][64 d], row-major (async global->LDS)
#pragma unroll
        for (int i = 0; i < 4; i++) {
            int c = tid + 128 * i;
            int row = c >> 3, cc = (c & 7) * 8;
            cp_b128_to_lds(&Ks[row * 72 + cc],
                           qkv + (rb + kbi * 64 + row) * NQKV + 1024 + h * 64 + cc);
        }
        // ---- stage V transposed: Vts[d][k] = V[k][d]
        {
            int k = tid >> 1, dbase = (tid & 1) * 32;
            const _Float16* vp = qkv + (rb + kbi * 64 + k) * NQKV + 2048 + h * 64 + dbase;
#pragma unroll
            for (int i = 0; i < 4; i++) {
                half8 v = *(const half8*)(vp + 8 * i);
#pragma unroll
                for (int e = 0; e < 8; e++)
                    Vts[(dbase + 8 * i + e) * 72 + k] = v[e];
            }
        }
        // ---- prefetch next key block's K/V rows into L2
        if (kbi < qb) {
            const _Float16* nk =
                qkv + (rb + (kbi + 1) * 64 + (tid >> 1)) * NQKV + 1024 + h * 64;
            __builtin_prefetch(nk, 0, 1);
            __builtin_prefetch(nk + 1024, 0, 1);   // matching V row
        }
        cp_async_wait();
        __syncthreads();

        // ---- S = Q * K^T  (4 n-tiles x 2 k-chunks = 8 WMMA)
        float8 s[4] = {};
#pragma unroll
        for (int jt = 0; jt < 4; jt++) {
#pragma unroll
            for (int f = 0; f < 2; f++) {
                int row = jt * 16 + (lane & 15);
                int kb = ((lane < 16) ? 0 : 8) + 32 * f;
                half8 lo = *(const half8*)(&Ks[row * 72 + kb]);
                half8 hi = *(const half8*)(&Ks[row * 72 + kb + 16]);
                half16 bk = __builtin_shufflevector(lo, hi, 0,1,2,3,4,5,6,7,8,9,10,11,12,13,14,15);
                s[jt] = __builtin_amdgcn_wmma_f32_16x16x32_f16(
                    false, aq[f], false, bk, (short)0, s[jt], false, false);
            }
        }

        // ---- causal mask (only the diagonal block needs it)
        if (kbi == qb) {
#pragma unroll
            for (int jt = 0; jt < 4; jt++) {
                int colg = jt * 16 + (lane & 15);
#pragma unroll
                for (int r = 0; r < 8; r++) {
                    int rowg = warp * 16 + ((lane < 16) ? r : r + 8);
                    if (colg > rowg) s[jt][r] = -1e30f;
                }
            }
        }

        // ---- online softmax update (row reductions across 16-lane halves)
#pragma unroll
        for (int r = 0; r < 8; r++) {
            float mx = s[0][r];
#pragma unroll
            for (int jt = 1; jt < 4; jt++) mx = fmaxf(mx, s[jt][r]);
#pragma unroll
            for (int off = 1; off < 16; off <<= 1)
                mx = fmaxf(mx, __shfl_xor(mx, off, 16));
            float mnew  = fmaxf(m[r], mx);
            float alpha = __expf(m[r] - mnew);
            m[r] = mnew;
            float rs = 0.0f;
#pragma unroll
            for (int jt = 0; jt < 4; jt++) {
                float p = __expf(s[jt][r] - mnew);
                s[jt][r] = p;
                rs += p;
            }
#pragma unroll
            for (int off = 1; off < 16; off <<= 1)
                rs += __shfl_xor(rs, off, 16);
            lsum[r] = lsum[r] * alpha + rs;
#pragma unroll
            for (int jt = 0; jt < 4; jt++) oacc[jt][r] *= alpha;
        }

        // ---- P: C-layout -> A-layout via per-wave LDS scratch
        const int pbase = warp * 16 * 72;
#pragma unroll
        for (int r = 0; r < 8; r++) {
            int rl = (lane < 16) ? r : r + 8;
#pragma unroll
            for (int jt = 0; jt < 4; jt++)
                Ps[pbase + rl * 72 + jt * 16 + (lane & 15)] = (_Float16)s[jt][r];
        }
        half16 ap[2];
#pragma unroll
        for (int f = 0; f < 2; f++) {
            int kb = ((lane < 16) ? 0 : 8) + 32 * f;
            const _Float16* pp = Ps + pbase + (lane & 15) * 72 + kb;
            half8 lo = *(const half8*)pp;
            half8 hi = *(const half8*)(pp + 16);
            ap[f] = __builtin_shufflevector(lo, hi, 0,1,2,3,4,5,6,7,8,9,10,11,12,13,14,15);
        }

        // ---- O += P * V  (4 d-tiles x 2 k-chunks = 8 WMMA)
#pragma unroll
        for (int jt = 0; jt < 4; jt++) {
#pragma unroll
            for (int f = 0; f < 2; f++) {
                int row = jt * 16 + (lane & 15);
                int kb = ((lane < 16) ? 0 : 8) + 32 * f;
                half8 lo = *(const half8*)(&Vts[row * 72 + kb]);
                half8 hi = *(const half8*)(&Vts[row * 72 + kb + 16]);
                half16 bv = __builtin_shufflevector(lo, hi, 0,1,2,3,4,5,6,7,8,9,10,11,12,13,14,15);
                oacc[jt] = __builtin_amdgcn_wmma_f32_16x16x32_f16(
                    false, ap[f], false, bv, (short)0, oacc[jt], false, false);
            }
        }
    }

    // ---- epilogue: O / l -> y[b,t, h*64+d] f16
    size_t orow = rb + (size_t)qb * 64 + warp * 16;
#pragma unroll
    for (int r = 0; r < 8; r++) {
        int rl = (lane < 16) ? r : r + 8;
        float inv = 1.0f / lsum[r];
#pragma unroll
        for (int jt = 0; jt < 4; jt++)
            y[(orow + rl) * D + h * 64 + jt * 16 + (lane & 15)] =
                (_Float16)(oacc[jt][r] * inv);
    }
}

// ---------------------------------------------------------------------------
extern "C" void kernel_launch(void* const* d_in, const int* in_sizes, int n_in,
                              void* d_out, int out_size, void* d_ws, size_t ws_size,
                              hipStream_t stream) {
    (void)in_sizes; (void)n_in; (void)out_size; (void)ws_size;
    const float* x      = (const float*)d_in[0];   // [B,T,D]
    const float* W_qkv  = (const float*)d_in[1];   // [D,3D]
    const float* W_proj = (const float*)d_in[2];   // [D,D]
    float* out = (float*)d_out;                    // [B,T,D]

    char* ws = (char*)d_ws;
    _Float16* xh     = (_Float16*)(ws);                             // 16 MB
    _Float16* wqkvt  = (_Float16*)(ws + (size_t)16 * 1024 * 1024);  // 6 MB  [3072][1024]
    _Float16* wprojt = (_Float16*)(ws + (size_t)22 * 1024 * 1024);  // 2 MB  [1024][1024]
    _Float16* qkvh   = (_Float16*)(ws + (size_t)24 * 1024 * 1024);  // 48 MB [8192][3072]
    _Float16* yh     = (_Float16*)(ws + (size_t)72 * 1024 * 1024);  // 16 MB [8192][1024]

    {
        int n = BT * D;
        cvt_f16_kernel<<<(n + 255) / 256, 256, 0, stream>>>(x, xh, n);
    }
    {
        int n = D * NQKV;
        cvtT_f16_kernel<<<(n + 255) / 256, 256, 0, stream>>>(W_qkv, wqkvt, D, NQKV);
    }
    {
        int n = D * D;
        cvtT_f16_kernel<<<(n + 255) / 256, 256, 0, stream>>>(W_proj, wprojt, D, D);
    }

    // QKV GEMM: [8192,1024] x [1024,3072] -> f16 [8192,3072]
    gemm_wmma_kernel<true><<<dim3(NQKV / 128, BT / 64), 256, 0, stream>>>(
        xh, wqkvt, (void*)qkvh, BT, NQKV, D);

    // causal flash attention -> f16 [8192,1024]
    attn_wmma_kernel<<<BSZ * H * (T / 64), 128, 0, stream>>>(qkvh, yh);

    // output projection: [8192,1024] x [1024,1024] -> f32 d_out
    gemm_wmma_kernel<false><<<dim3(D / 128, BT / 64), 256, 0, stream>>>(
        yh, wprojt, (void*)out, BT, D, D);
}
